// SGConvNet_23278722744787
// MI455X (gfx1250) — matrix operand order repaired and checked by reading, not compiled, not run on hardware
//
#include <hip/hip_runtime.h>
#include <hip/hip_bf16.h>

typedef __attribute__((ext_vector_type(16))) _Float16 v16h;
typedef __attribute__((ext_vector_type(8)))  float    v8f;

#define NGRAPH 64
#define NPG    128
#define WID    128
#define FIN    6
#define D2     768
#define KNN    100
#define SLOPE  0.01f
#define MVAL   (1.0f/101.0f)

// ---------------------------------------------------------------------------
// WMMA fragment loaders; A and B share this file's contiguous k order (not AMD's table order)
// A 16x32 f16: lane L -> M = L&15 ; VGPR v holds K pair:
//   k = (L>=16?16:0) + 2*v
// B 32x16 f16: lane L -> N = L&15 ; VGPR v holds K pair:
//   k = (L>=16?16:0) + 2*v
// C 16x16 f32: VGPR r -> M = r + (L>=16?8:0), N = L&15
// ---------------------------------------------------------------------------

static __device__ __forceinline__ v16h load_a_f16(const _Float16* A, int lda,
                                                  int row0, int k0, int lane) {
  v16h a;
  int m  = row0 + (lane & 15);
  int kb = (lane >= 16) ? 16 : 0;
#pragma unroll
  for (int r = 0; r < 8; ++r) {
    int k = k0 + kb + 2 * r;
    a[2 * r]     = A[m * lda + k];
    a[2 * r + 1] = A[m * lda + k + 1];
  }
  return a;
}

static __device__ __forceinline__ v16h load_a_f32(const float* A, int lda,
                                                  int row0, int k0, int lane) {
  v16h a;
  int m  = row0 + (lane & 15);
  int kb = (lane >= 16) ? 16 : 0;
#pragma unroll
  for (int r = 0; r < 8; ++r) {
    int k = k0 + kb + 2 * r;
    a[2 * r]     = (_Float16)A[m * lda + k];
    a[2 * r + 1] = (_Float16)A[m * lda + k + 1];
  }
  return a;
}

static __device__ __forceinline__ v16h load_b_f16(const _Float16* B, int ldb,
                                                  int k0, int col0, int lane) {
  v16h b;
  int n  = col0 + (lane & 15);
  int kg = k0 + ((lane >= 16) ? 16 : 0);
#pragma unroll
  for (int r = 0; r < 8; ++r) {
    b[2 * r]     = B[(kg + 2 * r)     * ldb + n];
    b[2 * r + 1] = B[(kg + 2 * r + 1) * ldb + n];
  }
  return b;
}

// Unguarded f32 global B loader (no divergent predication)
static __device__ __forceinline__ v16h load_b_f32(const float* B, int ldb,
                                                  int k0, int col0, int lane) {
  v16h b;
  int n  = col0 + (lane & 15);
  int kg = k0 + ((lane >= 16) ? 16 : 0);
#pragma unroll
  for (int r = 0; r < 8; ++r) {
    b[2 * r]     = (_Float16)B[(kg + 2 * r)     * ldb + n];
    b[2 * r + 1] = (_Float16)B[(kg + 2 * r + 1) * ldb + n];
  }
  return b;
}

static __device__ __forceinline__ void store_tile_f16(_Float16* D, int ldd,
                                                      int tm, int tn, v8f acc,
                                                      int lane) {
  int n  = tn * 16 + (lane & 15);
  int mo = (lane >= 16) ? 8 : 0;
#pragma unroll
  for (int r = 0; r < 8; ++r)
    D[(tm * 16 + r + mo) * ldd + n] = (_Float16)acc[r];
}

static __device__ __forceinline__ v8f wmma32(v16h a, v16h b, v8f c) {
  return __builtin_amdgcn_wmma_f32_16x16x32_f16(false, a, false, b,
                                                (short)0, c, false, false);
}

// ---------------------------------------------------------------------------
// kNN: per-graph 128x128 block. deg==101 uniformly => M[i][j] = 1/101 iff
// j in top-100-nearest(i) or j==i.  Exact top_k tie semantics via rank =
// #{k: d_k < d} + #{k<t: d_k == d}.
// ---------------------------------------------------------------------------
__global__ __launch_bounds__(128) void knn_kernel(const float* __restrict__ x,
                                                  _Float16* __restrict__ Mg) {
  __shared__ float xg[NPG * 8];
  __shared__ float sq[NPG];
  __shared__ float drow[NPG];
  int g = blockIdx.x, t = threadIdx.x;
  const float* xs = x + (size_t)g * NPG * FIN;
  float s = 0.f;
#pragma unroll
  for (int f = 0; f < FIN; ++f) { float v = xs[t * FIN + f]; xg[t * 8 + f] = v; s += v * v; }
  sq[t] = s;
  __syncthreads();
  _Float16* Mrow = Mg + (size_t)g * NPG * NPG;
  const _Float16 wv = (_Float16)MVAL;
  for (int i = 0; i < NPG; ++i) {
    float d;
    if (t == i) {
      d = 3.0e38f;  // masked diagonal
    } else {
      float dot = 0.f;
#pragma unroll
      for (int f = 0; f < FIN; ++f) dot += xg[i * 8 + f] * xg[t * 8 + f];
      d = sq[i] + sq[t] - 2.f * dot;
    }
    drow[t] = d;
    __syncthreads();
    int rank = 0;
    for (int k = 0; k < NPG; ++k) {
      float dk = drow[k];
      rank += (dk < d) || (dk == d && k < t);
    }
    Mrow[i * NPG + t] = (rank < KNN || t == i) ? wv : (_Float16)0.f;
    __syncthreads();
  }
}

// ---------------------------------------------------------------------------
// SGConv (K=2): out = leaky( M*(M*h) * W + b ), one WG per graph, 4 waves.
// LDS = 64KB: sM (M, later reused for W) + sT (h -> t1 -> t2), all f16.
// All three GEMMs run purely from LDS fragments (no divergent global loads).
// ---------------------------------------------------------------------------
template <int KIN, int KINPAD>
__global__ __launch_bounds__(128) void conv_kernel(const _Float16* __restrict__ Mg,
                                                   const float* __restrict__ hin,
                                                   const float* __restrict__ W,
                                                   const float* __restrict__ bias,
                                                   float* __restrict__ hout) {
  constexpr int NT1 = KINPAD / 16;   // tiles along feature dim of t1/t2
  constexpr int PW1 = (8 * NT1) / 4; // tiles per wave in propagation phases
  __shared__ _Float16 sM[NPG * NPG];       // 32 KB: M, then W (padded)
  __shared__ _Float16 sT[NPG * KINPAD];    // <=32 KB: h -> t1 -> t2

  int g = blockIdx.x, tid = threadIdx.x;
  int wave = tid >> 5, lane = tid & 31;

  // stage M
  const _Float16* Msrc = Mg + (size_t)g * NPG * NPG;
  for (int i = tid; i < NPG * NPG; i += 128) sM[i] = Msrc[i];

  // stage h zero-padded to KINPAD columns (uniform loops, no predication)
  const float* hg = hin + (size_t)g * NPG * KIN;
  if (KIN < KINPAD) {
    for (int i = tid; i < NPG * KINPAD; i += 128) sT[i] = (_Float16)0.f;
    __syncthreads();
    for (int i = tid; i < NPG * KIN; i += 128)
      sT[(i / KIN) * KINPAD + (i % KIN)] = (_Float16)hg[i];
  } else {
    for (int i = tid; i < NPG * KINPAD; i += 128) sT[i] = (_Float16)hg[i];
  }
  __syncthreads();

  // two propagation rounds: sT <- M @ sT (tiles in registers, then swap)
  for (int rep = 0; rep < 2; ++rep) {
    v8f acc[PW1];
#pragma unroll
    for (int t = 0; t < PW1; ++t) {
      int tile = wave + 4 * t;
      int tm = tile / NT1, tn = tile % NT1;
      v8f a_ = {};
      for (int k0 = 0; k0 < NPG; k0 += 32) {
        a_ = wmma32(load_a_f16(sM, NPG, tm * 16, k0, lane),
                    load_b_f16(sT, KINPAD, k0, tn * 16, lane), a_);
      }
      acc[t] = a_;
    }
    __syncthreads();
#pragma unroll
    for (int t = 0; t < PW1; ++t) {
      int tile = wave + 4 * t;
      store_tile_f16(sT, KINPAD, tile / NT1, tile % NT1, acc[t], lane);
    }
    __syncthreads();
  }

  // M is dead: stage W zero-padded into sM (KINPAD x WID)
  if (KIN < KINPAD) {
    for (int i = tid; i < KINPAD * WID; i += 128) sM[i] = (_Float16)0.f;
    __syncthreads();
    for (int i = tid; i < KIN * WID; i += 128) sM[i] = (_Float16)W[i];
  } else {
    for (int i = tid; i < KINPAD * WID; i += 128) sM[i] = (_Float16)W[i];
  }
  __syncthreads();

  // out = t2 @ W + b, leaky  (8x8 tiles, 16 per wave)
  for (int t = 0; t < 16; ++t) {
    int tile = wave + 4 * t;
    int tm = tile / 8, tn = tile % 8;
    v8f acc = {};
    for (int k0 = 0; k0 < KINPAD; k0 += 32)
      acc = wmma32(load_a_f16(sT, KINPAD, tm * 16, k0, lane),
                   load_b_f16(sM, WID, k0, tn * 16, lane), acc);
    int n  = tn * 16 + (lane & 15);
    int mo = (lane >= 16) ? 8 : 0;
    float bv = bias[n];
#pragma unroll
    for (int r = 0; r < 8; ++r) {
      float v = acc[r] + bv;
      v = (v > 0.f) ? v : SLOPE * v;
      hout[((size_t)g * NPG + tm * 16 + r + mo) * WID + n] = v;
    }
  }
}

// ---------------------------------------------------------------------------
// concat(mean, max) pooling per graph -> z[:, cofs .. cofs+255]
// ---------------------------------------------------------------------------
__global__ __launch_bounds__(128) void pool_kernel(const float* __restrict__ h,
                                                   float* __restrict__ z, int cofs) {
  int g = blockIdx.x, f = threadIdx.x;
  const float* hg = h + (size_t)g * NPG * WID;
  float s = 0.f, m = -3.4e38f;
  for (int n = 0; n < NPG; ++n) {
    float v = hg[n * WID + f];
    s += v;
    m = fmaxf(m, v);
  }
  z[g * D2 + cofs + f]       = s * (1.0f / NPG);
  z[g * D2 + cofs + WID + f] = m;
}

// ---------------------------------------------------------------------------
// BatchNorm1d over the 64-graph batch (biased variance)
// ---------------------------------------------------------------------------
__global__ __launch_bounds__(128) void bn_kernel(const float* __restrict__ z,
                                                 const float* __restrict__ gamma,
                                                 const float* __restrict__ beta,
                                                 float* __restrict__ zn) {
  int j = blockIdx.x * 128 + threadIdx.x;  // 0..767
  float m = 0.f;
  for (int r = 0; r < NGRAPH; ++r) m += z[r * D2 + j];
  m *= (1.0f / NGRAPH);
  float v = 0.f;
  for (int r = 0; r < NGRAPH; ++r) { float d = z[r * D2 + j] - m; v += d * d; }
  v *= (1.0f / NGRAPH);
  float inv = rsqrtf(v + 1e-5f) * gamma[j];
  float bt  = beta[j];
  for (int r = 0; r < NGRAPH; ++r) zn[r * D2 + j] = (z[r * D2 + j] - m) * inv + bt;
}

// ---------------------------------------------------------------------------
// Dense 64x768 @ 768x768 + b, leaky. One WG per 16-col tile; wave = M-tile.
// Unguarded direct-global f32->f16 fragments (problem sits in L2).
// ---------------------------------------------------------------------------
__global__ __launch_bounds__(128) void dense_kernel(const float* __restrict__ X,
                                                    const float* __restrict__ W,
                                                    const float* __restrict__ bias,
                                                    float* __restrict__ Y) {
  int tn = blockIdx.x;               // 0..47
  int wave = threadIdx.x >> 5, lane = threadIdx.x & 31;
  int tm = wave;                     // 64 rows = 4 tiles
  v8f acc = {};
  for (int k0 = 0; k0 < D2; k0 += 32) {
    v16h a = load_a_f32(X, D2, tm * 16, k0, lane);
    v16h b = load_b_f32(W, D2, k0, tn * 16, lane);
    acc = wmma32(a, b, acc);
  }
  int n  = tn * 16 + (lane & 15);
  int mo = (lane >= 16) ? 8 : 0;
  float bv = bias[n];
#pragma unroll
  for (int r = 0; r < 8; ++r) {
    float v = acc[r] + bv;
    v = (v > 0.f) ? v : SLOPE * v;
    Y[(tm * 16 + r + mo) * D2 + n] = v;
  }
}

// Output head: [64,768] @ [768,1] + b
__global__ __launch_bounds__(64) void out_kernel(const float* __restrict__ z,
                                                 const float* __restrict__ W,
                                                 const float* __restrict__ b,
                                                 float* __restrict__ out) {
  int g = threadIdx.x;
  float s = b[0];
  for (int k = 0; k < D2; ++k) s += z[g * D2 + k] * W[k];
  out[g] = s;
}

// ---------------------------------------------------------------------------
extern "C" void kernel_launch(void* const* d_in, const int* in_sizes, int n_in,
                              void* d_out, int out_size, void* d_ws, size_t ws_size,
                              hipStream_t stream) {
  (void)in_sizes; (void)n_in; (void)out_size; (void)ws_size;
  const float* x        = (const float*)d_in[0];
  // d_in[1] = batch (arange//128) -- structure is known, not needed
  const float* conv1_W  = (const float*)d_in[2];
  const float* conv1_b  = (const float*)d_in[3];
  const float* convs_W  = (const float*)d_in[4];
  const float* convs_b  = (const float*)d_in[5];
  const float* bn_gamma = (const float*)d_in[6];
  const float* bn_beta  = (const float*)d_in[7];
  const float* lin_W    = (const float*)d_in[8];
  const float* lin_b    = (const float*)d_in[9];
  const float* out_W    = (const float*)d_in[10];
  const float* out_b    = (const float*)d_in[11];

  char* ws = (char*)d_ws;
  _Float16* Mg = (_Float16*)ws;                               //  2 MB
  float* hA    = (float*)(ws + 2097152);                      //  4 MB
  float* hB    = (float*)(ws + 2097152 + 4194304);            //  4 MB
  float* z     = (float*)(ws + 2097152 + 2 * 4194304);        // 192 KB
  float* z1    = (float*)(ws + 2097152 + 2 * 4194304 + 196608);
  float* z2    = (float*)(ws + 2097152 + 2 * 4194304 + 2 * 196608);

  knn_kernel<<<NGRAPH, 128, 0, stream>>>(x, Mg);

  conv_kernel<FIN, 32><<<NGRAPH, 128, 0, stream>>>(Mg, x, conv1_W, conv1_b, hA);
  pool_kernel<<<NGRAPH, 128, 0, stream>>>(hA, z, 0);

  conv_kernel<WID, WID><<<NGRAPH, 128, 0, stream>>>(Mg, hA, convs_W, convs_b, hB);
  pool_kernel<<<NGRAPH, 128, 0, stream>>>(hB, z, 256);

  conv_kernel<WID, WID><<<NGRAPH, 128, 0, stream>>>(Mg, hB, convs_W + WID * WID,
                                                    convs_b + WID, hA);
  pool_kernel<<<NGRAPH, 128, 0, stream>>>(hA, z, 512);

  bn_kernel<<<6, 128, 0, stream>>>(z, bn_gamma, bn_beta, z1);

  const float* cur = z1;
  float* nxt = z2;
  for (int l = 0; l < 5; ++l) {
    dense_kernel<<<48, 128, 0, stream>>>(cur, lin_W + (size_t)l * D2 * D2,
                                         lin_b + l * D2, nxt);
    const float* tmp = nxt; nxt = (float*)cur; cur = tmp;
  }

  out_kernel<<<1, 64, 0, stream>>>(cur, out_W, out_b, (float*)d_out);
}
